// M_1803886265238
// MI455X (gfx1250) — matrix operand order repaired and checked by reading, not compiled, not run on hardware
//
#include <hip/hip_runtime.h>

// Problem sizes (exact-fit tiling)
#define ROWS 25088   // 128*14*14, = 196 * 128
#define CIN  384     // K,  = 12 * 32
#define FOUT 1536    // N,  = 12 * 128
#define BM   128
#define BN   128
#define BK   32
#define KSTEPS (CIN / BK)   // 12

typedef _Float16 h4   __attribute__((ext_vector_type(4)));
typedef _Float16 h8   __attribute__((ext_vector_type(8)));
typedef _Float16 v16h __attribute__((ext_vector_type(16)));
typedef float    f4   __attribute__((ext_vector_type(4)));
typedef float    v8f  __attribute__((ext_vector_type(8)));

// Async global->LDS copy, 16 bytes per lane, tracked by ASYNCcnt.
// vdst = 32-bit LDS byte-offset VGPR (AS3 pointer), vaddr = 64-bit global addr.
#define ASYNC_LOAD_B128(ldsptr, gptr)                                          \
    asm volatile("global_load_async_to_lds_b128 %0, %1, off"                   \
                 :: "v"((__attribute__((address_space(3))) void*)(ldsptr)),    \
                    "v"((const void*)(gptr))                                   \
                 : "memory")

__device__ __forceinline__ void wait_async0() {
#if __has_builtin(__builtin_amdgcn_s_wait_asynccnt)
    __builtin_amdgcn_s_wait_asynccnt(0);
#else
    asm volatile("s_wait_asynccnt 0x0" ::: "memory");
#endif
}

// ---------------------------------------------------------------------------
// Kernel 0: W (384 x 1536 f32, [k][n]) -> Wt (1536 x 384 f16, [n][k]).
// Transposed so GEMM B-fragments are contiguous along K. Reads coalesced in n.
// ---------------------------------------------------------------------------
__global__ void prep_w(const float* __restrict__ W, _Float16* __restrict__ Wt) {
    int idx = blockIdx.x * 256 + threadIdx.x;      // 0 .. 384*1536-1
    int n = idx % FOUT;
    int k = idx / FOUT;
    Wt[(size_t)n * CIN + k] = (_Float16)W[(size_t)k * FOUT + n];
}

// ---------------------------------------------------------------------------
// Kernel 1: LayerNorm over last dim (384), f32 in -> f16 out (workspace).
// One wave32 per row; 12 elements per lane; shuffle-based reduction.
// ---------------------------------------------------------------------------
__global__ void ln_kernel(const float* __restrict__ x,
                          const float* __restrict__ gamma,
                          const float* __restrict__ beta,
                          _Float16* __restrict__ xn) {
    const int wid  = threadIdx.x >> 5;
    const int lane = threadIdx.x & 31;
    const int row  = blockIdx.x * 8 + wid;

    const float* xr = x + (size_t)row * CIN;
    f4 v[3];
    float s = 0.0f, sq = 0.0f;
#pragma unroll
    for (int j = 0; j < 3; ++j) {
        v[j] = *(const f4*)(xr + lane * 4 + j * 128);
#pragma unroll
        for (int e = 0; e < 4; ++e) { float t = v[j][e]; s += t; sq += t * t; }
    }
#pragma unroll
    for (int m = 16; m >= 1; m >>= 1) {
        s  += __shfl_xor(s,  m, 32);
        sq += __shfl_xor(sq, m, 32);
    }
    const float mu  = s * (1.0f / CIN);
    const float var = sq * (1.0f / CIN) - mu * mu;
    const float rs  = rsqrtf(var + 1e-5f);

    _Float16* xo = xn + (size_t)row * CIN;
#pragma unroll
    for (int j = 0; j < 3; ++j) {
        const int col = lane * 4 + j * 128;
        h4 o;
#pragma unroll
        for (int e = 0; e < 4; ++e) {
            float t = (v[j][e] - mu) * rs * gamma[col + e] + beta[col + e];
            o[e] = (_Float16)t;
        }
        *(h4*)(xo + col) = o;
    }
}

// ---------------------------------------------------------------------------
// Kernel 2: GEMM (25088x384 f16) x (384x1536 f16, B from Wt[n][k]) + bias,
// exact GELU, f32 out. Block = 128x128, 8 waves as 2(M) x 4(N); each wave
// computes 4x2 tiles of 16x16 with v_wmma_f32_16x16x32_f16.
// K slabs staged to double-buffered LDS via global_load_async_to_lds_b128.
// ---------------------------------------------------------------------------
__global__ __launch_bounds__(256)
void gemm_kernel(const _Float16* __restrict__ xn,
                 const _Float16* __restrict__ Wt,
                 const float* __restrict__ bias,
                 float* __restrict__ out) {
    __shared__ __align__(16) _Float16 ldsA[2][BM * BK];  // [buf][m_local][k], row=64B
    __shared__ __align__(16) _Float16 ldsB[2][BN * BK];  // [buf][n_local][k], row=64B

    const int tid  = threadIdx.x;
    const int wid  = tid >> 5;
    const int lane = tid & 31;
    const int half = lane >> 4;     // 0/1: lane-group per ISA f16 WMMA layout
    const int lm   = lane & 15;
    const int wm   = wid >> 2;      // 0..1 -> 64 rows each
    const int wn   = wid & 3;       // 0..3 -> 32 cols each
    const int m0   = blockIdx.y * BM;
    const int n0   = blockIdx.x * BN;

    // Per-thread staging map: rows r0 and r0+64, one 16B (8 half) chunk each,
    // for both the A slab (128x32) and B slab (128x32): 4 async B128 per thread.
    const int r0 = tid >> 2;              // 0..63
    const int ch = (tid & 3) * 8;         // half-element column chunk

    auto issue = [&](int kb, int buf) {
        const int k0 = kb * BK;
        ASYNC_LOAD_B128(&ldsA[buf][(r0     ) * BK + ch],
                        xn + (size_t)(m0 + r0     ) * CIN + k0 + ch);
        ASYNC_LOAD_B128(&ldsA[buf][(r0 + 64) * BK + ch],
                        xn + (size_t)(m0 + r0 + 64) * CIN + k0 + ch);
        ASYNC_LOAD_B128(&ldsB[buf][(r0     ) * BK + ch],
                        Wt + (size_t)(n0 + r0     ) * CIN + k0 + ch);
        ASYNC_LOAD_B128(&ldsB[buf][(r0 + 64) * BK + ch],
                        Wt + (size_t)(n0 + r0 + 64) * CIN + k0 + ch);
    };

    v8f acc[4][2];
#pragma unroll
    for (int mt = 0; mt < 4; ++mt)
#pragma unroll
        for (int nt = 0; nt < 2; ++nt) acc[mt][nt] = 0.0f;

    issue(0, 0);   // prologue fill of buffer 0

    for (int kb = 0; kb < KSTEPS; ++kb) {
        const int cur = kb & 1;

        wait_async0();        // my fills of buf[cur] are in LDS
        __syncthreads();      // everyone's fills done; everyone's reads of
                              // buf[cur^1] (from kb-1) already locally complete

        if (kb + 1 < KSTEPS) issue(kb + 1, cur ^ 1);  // overlap next slab

        // A fragment: 16x32 f16. Lane: row lm; K = [half*8,+8) and [16+half*8,+8).
        v16h afr[4];
#pragma unroll
        for (int mt = 0; mt < 4; ++mt) {
            const _Float16* pa = &ldsA[cur][(wm * 64 + mt * 16 + lm) * BK];
            h8 lo = *(const h8*)(pa + half * 8);
            h8 hi = *(const h8*)(pa + 16 + half * 8);
            afr[mt] = __builtin_shufflevector(lo, hi, 0,1,2,3,4,5,6,7,8,9,10,11,12,13,14,15);
        }
        // B fragment: 32x16 f16. Lane: col lm; K = [half*8,+8) and [16+half*8,+8).
        v16h bfr[2];
#pragma unroll
        for (int nt = 0; nt < 2; ++nt) {
            const _Float16* pb = &ldsB[cur][(wn * 32 + nt * 16 + lm) * BK];
            h8 lo = *(const h8*)(pb + half * 8);
            h8 hi = *(const h8*)(pb + 16 + half * 8);
            bfr[nt] = __builtin_shufflevector(lo, hi, 0,1,2,3,4,5,6,7,8,9,10,11,12,13,14,15);
        }

#pragma unroll
        for (int mt = 0; mt < 4; ++mt)
#pragma unroll
            for (int nt = 0; nt < 2; ++nt)
                acc[mt][nt] = __builtin_amdgcn_wmma_f32_16x16x32_f16(
                    /*neg_a=*/false, afr[mt], /*neg_b=*/false, bfr[nt],
                    /*c_mod=*/(short)0, acc[mt][nt],
                    /*reuse_a=*/false, /*reuse_b=*/false);
    }

    // Epilogue: bias + exact GELU (erf form), f32 store.
    // C/D layout: VGPR e -> row = half*8 + e, col = lm.
#pragma unroll
    for (int nt = 0; nt < 2; ++nt) {
        const int ncol = n0 + wn * 32 + nt * 16 + lm;
        const float bv = bias[ncol];
#pragma unroll
        for (int mt = 0; mt < 4; ++mt) {
            const int rbase = m0 + wm * 64 + mt * 16 + half * 8;
#pragma unroll
            for (int e = 0; e < 8; ++e) {
                float y = acc[mt][nt][e] + bv;
                float g = 0.5f * y * (1.0f + erff(y * 0.70710678118654752f));
                out[(size_t)(rbase + e) * FOUT + ncol] = g;
            }
        }
    }
}

// ---------------------------------------------------------------------------
extern "C" void kernel_launch(void* const* d_in, const int* in_sizes, int n_in,
                              void* d_out, int out_size, void* d_ws, size_t ws_size,
                              hipStream_t stream) {
    const float* x     = (const float*)d_in[0];   // (128,14,14,384) f32
    const float* gamma = (const float*)d_in[1];   // (384,)
    const float* beta  = (const float*)d_in[2];   // (384,)
    const float* W     = (const float*)d_in[3];   // (384,1536)
    const float* b     = (const float*)d_in[4];   // (1536,)
    float* out = (float*)d_out;                   // (128,14,14,1536) f32

    // Workspace layout: [ Wt f16 : 1536*384 ][ xn f16 : 25088*384 ]
    const size_t WT_BYTES = (size_t)FOUT * CIN * sizeof(_Float16);  // 1,179,648
    _Float16* wt = (_Float16*)d_ws;
    _Float16* xn = (_Float16*)((char*)d_ws + WT_BYTES);

    prep_w<<<(FOUT * CIN) / 256, 256, 0, stream>>>(W, wt);
    ln_kernel<<<ROWS / 8, 256, 0, stream>>>(x, gamma, beta, xn);
    dim3 grid(FOUT / BN, ROWS / BM);   // (12, 196)
    gemm_kernel<<<grid, 256, 0, stream>>>(xn, wt, b, out);
}
